// PointTransformerLayer_29480655520247
// MI455X (gfx1250) — compile-verified, matrix operand order checked by reading, not run on hardware
//
#include <hip/hip_runtime.h>
#include <hip/hip_fp16.h>

typedef __attribute__((ext_vector_type(16))) _Float16 v16h;
typedef __attribute__((ext_vector_type(8)))  float    v8f;

#define BB   2
#define NNp  16384
#define KKn  16
#define CIN  64
#define COUT 64
#define ATT  8
#define NPTS (BB*NNp)          /* 32768 */
#define NNBR (NPTS*KKn)        /* 524288 */
#define CNTF ((float)NNBR)
#define EPSV 1e-5f

/* workspace layout (float offsets) */
#define WS_Q   0
#define WS_KF  (WS_Q  + NPTS*COUT)     /* 2097152 */
#define WS_VF  (WS_KF + NPTS*COUT)
#define WS_D1  (WS_VF + NPTS*COUT)
#define WS_H1  (WS_D1 + NNBR*3)
#define WS_ST  (WS_H1 + NNBR*ATT)
/* stats: [0..2] d_sum, [3..5] d_sq, [8..71] g1_sum, [72..135] g1_sq,
          [136..143] g2_sum, [144..151] g2_sq */

__global__ void zero_stats(float* __restrict__ st) {
    st[threadIdx.x] = 0.f;
}

/* ---- Pass 1: q/kf/vf = feat @ W + b via WMMA f16->f32 -------------------- */
__global__ void qkv_wmma(const float* __restrict__ X,
                         const float* __restrict__ Wq, const float* __restrict__ bq,
                         const float* __restrict__ Wk, const float* __restrict__ bk,
                         const float* __restrict__ Wv, const float* __restrict__ bv,
                         float* __restrict__ q, float* __restrict__ kf, float* __restrict__ vf) {
    int wave = (blockIdx.x * blockDim.x + threadIdx.x) >> 5;
    int lane = threadIdx.x & 31;
    const int tilesPerMat = (NPTS/16) * 4;          /* 8192 */
    int mat = wave / tilesPerMat;
    int rem = wave % tilesPerMat;
    int tm = rem >> 2;
    int tn = rem & 3;
    const float *W, *bias; float *out;
    if      (mat == 0) { W = Wq; bias = bq; out = q;  }
    else if (mat == 1) { W = Wk; bias = bk; out = kf; }
    else               { W = Wv; bias = bv; out = vf; }

    int mrow = tm*16 + (lane & 15);
    int ncol = tn*16 + (lane & 15);
    int kb   = (lane >> 4) * 8;   /* K sub-base per ISA 16-bit A/B layout */

    v8f c = {};
    #pragma unroll
    for (int k0 = 0; k0 < 64; k0 += 32) {
        v16h a, b;
        const float* xr = X + mrow*CIN + k0 + kb;
        #pragma unroll
        for (int i = 0; i < 8; ++i) {
            a[i]   = (_Float16)xr[i];
            a[i+8] = (_Float16)xr[16+i];
        }
        const float* wr = W + (k0 + kb)*COUT + ncol;
        #pragma unroll
        for (int i = 0; i < 8; ++i) {
            b[i]   = (_Float16)wr[i*COUT];
            b[i+8] = (_Float16)wr[(16+i)*COUT];
        }
        c = __builtin_amdgcn_wmma_f32_16x16x32_f16(false, a, false, b, (short)0, c, false, false);
    }
    float bv_ = bias[ncol];
    #pragma unroll
    for (int r = 0; r < 8; ++r) {
        int m = tm*16 + r + ((lane >> 4) << 3);
        out[m*COUT + ncol] = c[r] + bv_;
    }
}

/* ---- Pass 2: d1 = rel @ Wd1 + bd1; accumulate 3-ch BN stats -------------- */
__global__ void dstats_kernel(const float* __restrict__ xyz, const int* __restrict__ knn,
                              const float* __restrict__ Wd1, const float* __restrict__ bd1,
                              float* __restrict__ d1, float* __restrict__ st) {
    __shared__ float s[6];
    int t = threadIdx.x;
    if (t < 6) s[t] = 0.f;
    __syncthreads();
    int nbr = blockIdx.x * blockDim.x + t;
    int p = nbr / KKn;
    int b = p / NNp;
    int g = b*NNp + knn[nbr];
    float rel[3];
    #pragma unroll
    for (int i = 0; i < 3; ++i) rel[i] = xyz[p*3+i] - xyz[g*3+i];
    #pragma unroll
    for (int j = 0; j < 3; ++j) {
        float a = bd1[j];
        #pragma unroll
        for (int i = 0; i < 3; ++i) a += rel[i]*Wd1[i*3+j];
        d1[nbr*3+j] = a;
        atomicAdd(&s[j],   a);
        atomicAdd(&s[3+j], a*a);
    }
    __syncthreads();
    if (t < 6) atomicAdd(&st[t], s[t]);
}

/* helper: BN(d1)+ReLU scales for pos_enc */
__device__ __forceinline__ void d_bn_coeffs(const float* st, const float* gd, float* md, float* ivd) {
    #pragma unroll
    for (int j = 0; j < 3; ++j) {
        float m = st[j] / CNTF;
        float v = st[3+j] / CNTF - m*m;
        md[j]  = m;
        ivd[j] = rsqrtf(v + EPSV) * gd[j];
    }
}

/* ---- Pass 3: accumulate 64-ch BN stats of attn_in (recomputed) ----------- */
__global__ void g1stats_kernel(const float* __restrict__ qf, const float* __restrict__ kf,
                               const int* __restrict__ knn, const float* __restrict__ d1,
                               const float* __restrict__ Wd2, const float* __restrict__ bd2,
                               const float* __restrict__ gd, const float* __restrict__ bed,
                               float* __restrict__ st) {
    int t = threadIdx.x;
    int c = t & 63, slot = t >> 6;
    float md[3], ivd[3];
    d_bn_coeffs(st, gd, md, ivd);
    float be0 = bed[0], be1 = bed[1], be2 = bed[2];
    float w0 = Wd2[c], w1 = Wd2[64+c], w2 = Wd2[128+c], b2 = bd2[c];
    float sum = 0.f, sq = 0.f;
    const int per = NNBR / gridDim.x;
    int base = blockIdx.x * per;
    for (int it = slot; it < per; it += 4) {
        int nbr = base + it;
        int p = nbr / KKn;
        int b = p / NNp;
        int g = b*NNp + knn[nbr];
        float t0 = fmaxf((d1[nbr*3+0]-md[0])*ivd[0] + be0, 0.f);
        float t1 = fmaxf((d1[nbr*3+1]-md[1])*ivd[1] + be1, 0.f);
        float t2 = fmaxf((d1[nbr*3+2]-md[2])*ivd[2] + be2, 0.f);
        float pe = b2 + t0*w0 + t1*w1 + t2*w2;
        float x = qf[p*64+c] - kf[g*64+c] + pe;
        sum += x; sq += x*x;
    }
    __shared__ float sm[256];
    sm[t] = sum; __syncthreads();
    if (t < 64) atomicAdd(&st[8+t],  sm[t]+sm[t+64]+sm[t+128]+sm[t+192]);
    __syncthreads();
    sm[t] = sq;  __syncthreads();
    if (t < 64) atomicAdd(&st[72+t], sm[t]+sm[t+64]+sm[t+128]+sm[t+192]);
}

/* ---- Pass 4: h1 = relu(bn_g1(attn_in)) @ Wg1 + bg1; 8-ch stats ----------- */
__global__ void h1_kernel(const float* __restrict__ qf, const float* __restrict__ kf,
                          const int* __restrict__ knn, const float* __restrict__ d1,
                          const float* __restrict__ Wd2, const float* __restrict__ bd2,
                          const float* __restrict__ gd, const float* __restrict__ bed,
                          const float* __restrict__ gg1, const float* __restrict__ beg1,
                          const float* __restrict__ Wg1, const float* __restrict__ bg1,
                          float* __restrict__ h1, float* __restrict__ st) {
    int t = threadIdx.x;
    int c = t & 63, slot = t >> 6;
    __shared__ float sWg1[512];
    __shared__ float sY[256];
    sWg1[t] = Wg1[t]; sWg1[256+t] = Wg1[256+t];
    float md[3], ivd[3];
    d_bn_coeffs(st, gd, md, ivd);
    float be0 = bed[0], be1 = bed[1], be2 = bed[2];
    float m1 = st[8+c] / CNTF;
    float v1 = st[72+c] / CNTF - m1*m1;
    float iv1 = rsqrtf(v1 + EPSV) * gg1[c];
    float bb1 = beg1[c];
    float w0 = Wd2[c], w1 = Wd2[64+c], w2 = Wd2[128+c], b2 = bd2[c];
    float gs = 0.f, gq = 0.f;
    const int per = NNBR / gridDim.x;
    int base = blockIdx.x * per;
    __syncthreads();
    for (int it = slot; it < per; it += 4) {
        int nbr = base + it;
        int p = nbr / KKn;
        int b = p / NNp;
        int g = b*NNp + knn[nbr];
        float t0 = fmaxf((d1[nbr*3+0]-md[0])*ivd[0] + be0, 0.f);
        float t1 = fmaxf((d1[nbr*3+1]-md[1])*ivd[1] + be1, 0.f);
        float t2 = fmaxf((d1[nbr*3+2]-md[2])*ivd[2] + be2, 0.f);
        float pe = b2 + t0*w0 + t1*w1 + t2*w2;
        float x = qf[p*64+c] - kf[g*64+c] + pe;
        sY[slot*64+c] = fmaxf((x - m1)*iv1 + bb1, 0.f);
        __syncthreads();
        if (c < 8) {
            float acc = bg1[c];
            #pragma unroll
            for (int cc = 0; cc < 64; ++cc) acc += sY[slot*64+cc]*sWg1[cc*8+c];
            h1[nbr*8+c] = acc;
            gs += acc; gq += acc*acc;
        }
        __syncthreads();
    }
    if (c < 8) { sY[slot*8+c] = gs; sY[32 + slot*8+c] = gq; }
    __syncthreads();
    if (t < 8) {
        atomicAdd(&st[136+t], sY[t]+sY[8+t]+sY[16+t]+sY[24+t]);
        atomicAdd(&st[144+t], sY[32+t]+sY[40+t]+sY[48+t]+sY[56+t]);
    }
}

/* ---- Pass 5: h2, softmax over K, aggregate (v+pos_enc) ------------------- */
__global__ void out_kernel(const float* __restrict__ vf, const int* __restrict__ knn,
                           const float* __restrict__ d1, const float* __restrict__ h1,
                           const float* __restrict__ Wd2, const float* __restrict__ bd2,
                           const float* __restrict__ gd, const float* __restrict__ bed,
                           const float* __restrict__ gg2, const float* __restrict__ beg2,
                           const float* __restrict__ Wg2, const float* __restrict__ bg2,
                           const float* __restrict__ st, float* __restrict__ out) {
    int t = threadIdx.x;
    int c = t & 63, ps = t >> 6;
    int p = blockIdx.x*4 + ps;
    int b = p / NNp;
    int a = c & 7;
    float md[3], ivd[3];
    d_bn_coeffs(st, gd, md, ivd);
    float be0 = bed[0], be1 = bed[1], be2 = bed[2];
    float m2[8], iv2[8], be2j[8], wg2[8];
    #pragma unroll
    for (int j = 0; j < 8; ++j) {
        float m = st[136+j] / CNTF;
        float v = st[144+j] / CNTF - m*m;
        m2[j]  = m;
        iv2[j] = rsqrtf(v + EPSV) * gg2[j];
        be2j[j] = beg2[j];
        wg2[j] = Wg2[j*8+a];
    }
    float bg2a = bg2[a];
    float ek[16], hmax = -1e30f;
    #pragma unroll
    for (int k = 0; k < 16; ++k) {
        int nbr = p*16 + k;
        float acc = bg2a;
        #pragma unroll
        for (int j = 0; j < 8; ++j) {
            float y = fmaxf((h1[nbr*8+j] - m2[j])*iv2[j] + be2j[j], 0.f);
            acc += y*wg2[j];
        }
        ek[k] = acc;
        hmax = fmaxf(hmax, acc);
    }
    float ssum = 0.f;
    #pragma unroll
    for (int k = 0; k < 16; ++k) { ek[k] = __expf(ek[k] - hmax); ssum += ek[k]; }
    float inv = 1.f / ssum;
    float w0 = Wd2[c], w1 = Wd2[64+c], w2 = Wd2[128+c], b2 = bd2[c];
    float res = 0.f;
    #pragma unroll
    for (int k = 0; k < 16; ++k) {
        int nbr = p*16 + k;
        int g = b*NNp + knn[nbr];
        float t0 = fmaxf((d1[nbr*3+0]-md[0])*ivd[0] + be0, 0.f);
        float t1 = fmaxf((d1[nbr*3+1]-md[1])*ivd[1] + be1, 0.f);
        float t2 = fmaxf((d1[nbr*3+2]-md[2])*ivd[2] + be2, 0.f);
        float pe = b2 + t0*w0 + t1*w1 + t2*w2;
        res += (ek[k]*inv) * (vf[g*64+c] + pe);
    }
    out[p*64+c] = res;
}

extern "C" void kernel_launch(void* const* d_in, const int* in_sizes, int n_in,
                              void* d_out, int out_size, void* d_ws, size_t ws_size,
                              hipStream_t stream) {
    (void)in_sizes; (void)n_in; (void)out_size; (void)ws_size;
    const float* xyz  = (const float*)d_in[0];
    const float* feat = (const float*)d_in[1];
    const int*   knn  = (const int*)  d_in[2];
    const float* Wq = (const float*)d_in[3];  const float* bq = (const float*)d_in[4];
    const float* Wk = (const float*)d_in[5];  const float* bk = (const float*)d_in[6];
    const float* Wv = (const float*)d_in[7];  const float* bv = (const float*)d_in[8];
    const float* Wd1 = (const float*)d_in[9];  const float* bd1 = (const float*)d_in[10];
    const float* gd  = (const float*)d_in[11]; const float* bed = (const float*)d_in[12];
    const float* Wd2 = (const float*)d_in[13]; const float* bd2 = (const float*)d_in[14];
    const float* gg1 = (const float*)d_in[15]; const float* beg1 = (const float*)d_in[16];
    const float* Wg1 = (const float*)d_in[17]; const float* bg1  = (const float*)d_in[18];
    const float* gg2 = (const float*)d_in[19]; const float* beg2 = (const float*)d_in[20];
    const float* Wg2 = (const float*)d_in[21]; const float* bg2  = (const float*)d_in[22];

    float* ws = (float*)d_ws;
    float* q  = ws + WS_Q;
    float* kf = ws + WS_KF;
    float* vf = ws + WS_VF;
    float* d1 = ws + WS_D1;
    float* h1 = ws + WS_H1;
    float* st = ws + WS_ST;

    zero_stats<<<dim3(1), dim3(256), 0, stream>>>(st);
    qkv_wmma<<<dim3(3*(NPTS/16)*4/8), dim3(256), 0, stream>>>(feat, Wq, bq, Wk, bk, Wv, bv, q, kf, vf);
    dstats_kernel<<<dim3(NNBR/256), dim3(256), 0, stream>>>(xyz, knn, Wd1, bd1, d1, st);
    g1stats_kernel<<<dim3(1024), dim3(256), 0, stream>>>(q, kf, knn, d1, Wd2, bd2, gd, bed, st);
    h1_kernel<<<dim3(2048), dim3(256), 0, stream>>>(q, kf, knn, d1, Wd2, bd2, gd, bed,
                                                    gg1, beg1, Wg1, bg1, h1, st);
    out_kernel<<<dim3(NPTS/4), dim3(256), 0, stream>>>(vf, knn, d1, h1, Wd2, bd2, gd, bed,
                                                       gg2, beg2, Wg2, bg2, st, (float*)d_out);
}
